// blockSAGEsq_11527692222848
// MI455X (gfx1250) — matrix operand order repaired and checked by reading, not compiled, not run on hardware
//
#include <hip/hip_runtime.h>
#include <hip/hip_bf16.h>
#include <cstdint>

#define N_NODES 50000
#define N_EDGES 600000
#define DIMS    128

typedef __attribute__((ext_vector_type(16))) _Float16 v16h;
typedef __attribute__((ext_vector_type(8)))  float    v8f;
typedef __attribute__((ext_vector_type(4)))  float    v4f;

// ---------------------------------------------------------------- zero
__global__ void zero_f32(float* __restrict__ p, int n) {
  int i = blockIdx.x * blockDim.x + threadIdx.x;
  if (i < n) p[i] = 0.0f;
}

// ------------------------------------------- Bt[j][k] = f16([W_l | W_r])
// Bt is [128][256] f16, row j = output column, col k = reduction index.
__global__ void wprep(const float* __restrict__ Wl, const float* __restrict__ Wr,
                      _Float16* __restrict__ Bt) {
  int t = blockIdx.x * blockDim.x + threadIdx.x;
  if (t >= DIMS * 2 * DIMS) return;
  int j = t >> 8;        // 0..127
  int k = t & 255;       // 0..255
  float v = (k < DIMS) ? Wl[j * DIMS + k] : Wr[j * DIMS + (k - DIMS)];
  Bt[t] = (_Float16)v;
}

// ------------------------------------ scatter-add of x[src] onto dst rows
// One thread per (edge, 4-float chunk): 32 chunks cover D=128.
__global__ void scatter_mean_acc(const float* __restrict__ x, const int* __restrict__ ei,
                                 float* __restrict__ aggsum, float* __restrict__ counts) {
  int t = blockIdx.x * blockDim.x + threadIdx.x;
  if (t >= N_EDGES * 32) return;
  int e = t >> 5;
  int c = t & 31;
  int s = ei[e];
  int d = ei[N_EDGES + e];
  const float4 v = ((const float4*)(x + (size_t)s * DIMS))[c];
  float* dp = aggsum + (size_t)d * DIMS + c * 4;
  atomicAdd(dp + 0, v.x);
  atomicAdd(dp + 1, v.y);
  atomicAdd(dp + 2, v.z);
  atomicAdd(dp + 3, v.w);
  if (c == 0) atomicAdd(counts + d, 1.0f);
}

// ------------------------------------------------- agg = sum / max(cnt,1)
__global__ void finalize_mean(float* __restrict__ aggsum, const float* __restrict__ counts) {
  int i = blockIdx.x * blockDim.x + threadIdx.x;
  if (i >= N_NODES * DIMS) return;
  float c = counts[i >> 7];
  aggsum[i] *= 1.0f / fmaxf(c, 1.0f);
}

// --------------------------- out = elu(agg@Wl^T + x@Wr^T + b) via WMMA f16
// One wave per 16-row tile; 8 v8f accumulators cover all 128 output cols.
__global__ __launch_bounds__(256) void sage_gemm_wmma(
    const float* __restrict__ agg, const float* __restrict__ x,
    const _Float16* __restrict__ Bt, const float* __restrict__ bias,
    float* __restrict__ out) {
  const int lane = threadIdx.x & 31;
  const int wave = threadIdx.x >> 5;
  const int tile = blockIdx.x * 8 + wave;
  if (tile >= N_NODES / 16) return;   // wave-uniform: EXEC stays all-1s for WMMA

  const int m    = lane & 15;   // A row within tile / B column
  const int half = lane >> 4;   // lane group 0/1
  const int n0   = tile * 16;

  v8f acc[8] = {};

  const float* Asrc0 = agg + (size_t)(n0 + m) * DIMS;
  const float* Asrc1 = x   + (size_t)(n0 + m) * DIMS;

#pragma unroll
  for (int s = 0; s < 2; ++s) {
    const float* A = s ? Asrc1 : Asrc0;
#pragma unroll
    for (int kb = 0; kb < DIMS; kb += 32) {
      // A layout (matches B): lane group 0 holds K = kb+[0..15];
      // lane group 1 holds K = kb+16+[0..15]. One contiguous
      // 16-float chunk per lane, converted f32->f16 in registers.
      float af[16];
      *(v4f*)(af + 0)  = *(const v4f*)(A + kb + 16 * half);
      *(v4f*)(af + 4)  = *(const v4f*)(A + kb + 16 * half + 4);
      *(v4f*)(af + 8)  = *(const v4f*)(A + kb + 16 * half + 8);
      *(v4f*)(af + 12) = *(const v4f*)(A + kb + 16 * half + 12);
      v16h a;
#pragma unroll
      for (int i = 0; i < 16; ++i) a[i] = (_Float16)af[i];

      // B layout (same contiguous k order as A): element i -> K = kB + i, N = m; 32 contiguous bytes/lane.
      const int kB = s * DIMS + kb + 16 * half;
#pragma unroll
      for (int jt = 0; jt < 8; ++jt) {
        const v16h b = *(const v16h*)(Bt + (size_t)(jt * 16 + m) * (2 * DIMS) + kB);
        acc[jt] = __builtin_amdgcn_wmma_f32_16x16x32_f16(
            /*neg_a=*/false, a, /*neg_b=*/false, b,
            /*c_mod=*/(short)0, acc[jt], /*reuse_a=*/false, /*reuse_b=*/false);
      }
    }
  }

  // Epilogue: bias + ELU, store per C/D layout (row = v + 8*half, col = m).
#pragma unroll
  for (int jt = 0; jt < 8; ++jt) {
    const int j = jt * 16 + m;
    const float bb = bias[j];
#pragma unroll
    for (int v = 0; v < 8; ++v) {
      float z = acc[jt][v] + bb;
      z = (z > 0.0f) ? z : (__expf(z) - 1.0f);
      out[(size_t)(n0 + v + 8 * half) * DIMS + j] = z;
    }
  }
}

// ---------------------------------------------------------------- launch
extern "C" void kernel_launch(void* const* d_in, const int* in_sizes, int n_in,
                              void* d_out, int out_size, void* d_ws, size_t ws_size,
                              hipStream_t stream) {
  const float* x  = (const float*)d_in[0];
  const int*   ei = (const int*)d_in[1];
  const float* Wl = (const float*)d_in[2];
  const float* bl = (const float*)d_in[3];
  const float* Wr = (const float*)d_in[4];
  float* out = (float*)d_out;

  char* ws = (char*)d_ws;
  float*    aggsum = (float*)ws;                                         // 25,600,000 B
  float*    counts = (float*)(ws + (size_t)N_NODES * DIMS * 4);          //    200,000 B
  _Float16* Bt     = (_Float16*)(ws + (size_t)N_NODES * DIMS * 4 + 204800); // 65,536 B

  // 1) zero agg_sum + counts (contiguous region)
  {
    int n = N_NODES * DIMS + N_NODES;
    zero_f32<<<(n + 255) / 256, 256, 0, stream>>>(aggsum, n);
  }
  // 2) weight prep (f16 B matrix, [j][k] layout)
  wprep<<<(2 * DIMS * DIMS + 255) / 256, 256, 0, stream>>>(Wl, Wr, Bt);
  // 3) scatter-add messages + counts
  scatter_mean_acc<<<(N_EDGES * 32) / 256, 256, 0, stream>>>(x, ei, aggsum, counts);
  // 4) mean
  finalize_mean<<<(N_NODES * DIMS + 255) / 256, 256, 0, stream>>>(aggsum, counts);
  // 5) fused dual-GEMM + bias + ELU via WMMA
  sage_gemm_wmma<<<(N_NODES / 16 + 7) / 8, 256, 0, stream>>>(aggsum, x, Bt, bl, out);

  // 6) second tuple output: edge_index passthrough into the tail of d_out.
  // Byte count derived from out_size so it is index-dtype agnostic.
  long long tail = (long long)out_size - (long long)N_NODES * DIMS;
  if (tail > 0) {
    hipMemcpyAsync(out + (size_t)N_NODES * DIMS, d_in[1],
                   (size_t)tail * sizeof(float), hipMemcpyDeviceToDevice, stream);
  }
}
